// QJLSketch_80676665688761
// MI455X (gfx1250) — compile-verified, matrix operand order checked by reading, not run on hardware
//
#include <hip/hip_runtime.h>

// ---- CDNA5 WMMA vector types ----
typedef __attribute__((ext_vector_type(16))) _Float16 v16h;
typedef __attribute__((ext_vector_type(16))) __bf16   v16bf;
typedef __attribute__((ext_vector_type(8)))  float    v8f;

#define HEADS     128      // B*H = 4*32
#define HD        128      // head dim
#define SD        256      // sketch dim
#define QLEN      16
#define KLEN      8192
#define KC        256      // k rows per chunk (16 tiles -> one per wave)
#define PARTS     4        // workgroups per head
#define CHUNKS    (KLEN / (PARTS * KC))   // 8
#define GT_STRIDE 130      // padded u16 row stride for Gt (bank spread, 4B-aligned pairs)

__device__ __forceinline__ unsigned short bf16_bits(float x) {
    union { float f; unsigned int u; } a; a.f = x;
    unsigned int u = a.u;
    unsigned int r = (u + 0x7FFFu + ((u >> 16) & 1u)) >> 16;   // RNE (used only in one-time G staging)
    return (unsigned short)r;
}
__device__ __forceinline__ float bf16_f32(unsigned short b) {
    union { unsigned int u; float f; } a; a.u = ((unsigned int)b) << 16;
    return a.f;
}
__device__ __forceinline__ unsigned short f16_bits(float x) {
    union { _Float16 h[2]; unsigned int u; } p;
    p.h[0] = (_Float16)x; p.h[1] = (_Float16)0.f;
    return (unsigned short)(p.u & 0xFFFFu);
}
// K-pair base inside a 32-wide K step for 16-bit A/B fragment layout (ISA 7.12.2):
// lanes 0-15 (h=0): VGPR0..3 -> K 0..7, VGPR4..7 -> K 16..23 ; lanes 16-31 (h=1): +8
__device__ __forceinline__ int kpos(int v, int h) {
    return (v < 4 ? 2 * v : 16 + 2 * (v - 4)) + 8 * h;
}

// Load one lane's half-row as 16 consecutive-float4 pieces (two 8-float runs per 32-wide step).
__device__ __forceinline__ void load_raw_row(const float* __restrict__ row, int half, float4* raw) {
    #pragma unroll
    for (int st = 0; st < 4; ++st)
        #pragma unroll
        for (int g = 0; g < 2; ++g) {
            const int base = 32 * st + 16 * g + 8 * half;
            raw[(st * 2 + g) * 2 + 0] = *(const float4*)(row + base);
            raw[(st * 2 + g) * 2 + 1] = *(const float4*)(row + base + 4);
        }
}
// Convert raw float4s -> bf16 hi/lo A-fragments using exact truncation split:
// hi = bits&0xffff0000 (exact), lo = x - hi (exact); dropped lo*lo term ~2^-16 rel.
__device__ __forceinline__ void cvt_raw(const float4* raw, unsigned int Ahi[4][8], unsigned int Alo[4][8]) {
    #pragma unroll
    for (int st = 0; st < 4; ++st)
        #pragma unroll
        for (int g = 0; g < 2; ++g) {
            const float4 r0 = raw[(st * 2 + g) * 2 + 0];
            const float4 r1 = raw[(st * 2 + g) * 2 + 1];
            const float f[8] = { r0.x, r0.y, r0.z, r0.w, r1.x, r1.y, r1.z, r1.w };
            #pragma unroll
            for (int j = 0; j < 4; ++j) {
                const int v = 4 * g + j;
                union { float f; unsigned int u; } x0, x1, h0, h1, l0, l1;
                x0.f = f[2 * j]; x1.f = f[2 * j + 1];
                h0.u = x0.u & 0xffff0000u;      // hi as float (truncated bf16)
                h1.u = x1.u & 0xffff0000u;
                l0.f = x0.f - h0.f;             // exact residual
                l1.f = x1.f - h1.f;
                Ahi[st][v] = (x0.u >> 16) | (x1.u & 0xffff0000u);   // packed bf16 pair (v_perm)
                Alo[st][v] = (l0.u >> 16) | (l1.u & 0xffff0000u);   // truncated lo pair
            }
        }
}

__global__ __launch_bounds__(512, 1)
void qjl_sketch_kernel(const float* __restrict__ q, const float* __restrict__ k,
                       const float* __restrict__ G, float* __restrict__ out)
{
    __shared__ unsigned short sGthi[SD * GT_STRIDE];            // 65 KB
    __shared__ unsigned short sGtlo[SD * GT_STRIDE];            // 65 KB
    __shared__ unsigned int   sBfrag[(KC / 16) * 8 * 32 * 8];   // 128 KB  [kt][t][lane][v]
    __shared__ unsigned int   sAfrag[8 * 32 * 8];               // 8 KB    [t][lane][v]

    const int tid  = threadIdx.x;
    const int lane = tid & 31;
    const int wave = tid >> 5;          // 0..15
    const int m16  = lane & 15;
    const int half = lane >> 4;

    const int head = blockIdx.x >> 2;   // 0..127
    const int part = blockIdx.x & 3;

    // ---------- Phase 0a: G -> LDS, transposed, split into bf16 hi/lo (RNE, one-time) ----------
    {
        const int s  = tid & 255;
        const int dh = tid >> 8;        // 0..1
        for (int j = 0; j < HD / 2; ++j) {
            int d = dh * (HD / 2) + j;
            float g = G[d * SD + s];
            unsigned short hb = bf16_bits(g);
            float lo = g - bf16_f32(hb);
            sGthi[s * GT_STRIDE + d] = hb;
            sGtlo[s * GT_STRIDE + d] = bf16_bits(lo);
        }
    }
    __syncthreads();

    // ---------- Phase 0b: q_sketch via 3-term bf16 WMMA; scatter f16 into A-fragments ----------
    {
        float4 raw[16];
        unsigned int Qhi[4][8], Qlo[4][8];
        load_raw_row(q + ((size_t)head * QLEN + m16) * HD, half, raw);
        cvt_raw(raw, Qhi, Qlo);

        const int scol = wave * 16 + m16;   // this wave computes s-tile = wave
        v8f a0 = {}, a1 = {};
        #pragma unroll
        for (int st = 0; st < 4; ++st) {
            union { unsigned int u[8]; v16bf v; } bh, bl, ah, al;
            #pragma unroll
            for (int v = 0; v < 8; ++v) {
                int d = 32 * st + kpos(v, half);
                bh.u[v] = *(const unsigned int*)(sGthi + scol * GT_STRIDE + d);
                bl.u[v] = *(const unsigned int*)(sGtlo + scol * GT_STRIDE + d);
                ah.u[v] = Qhi[st][v];
                al.u[v] = Qlo[st][v];
            }
            v8f& acc = (st & 1) ? a1 : a0;
            acc = __builtin_amdgcn_wmma_f32_16x16x32_bf16(false, ah.v, false, bh.v, (short)0, acc, false, false);
            acc = __builtin_amdgcn_wmma_f32_16x16x32_bf16(false, ah.v, false, bl.v, (short)0, acc, false, false);
            acc = __builtin_amdgcn_wmma_f32_16x16x32_bf16(false, al.v, false, bh.v, (short)0, acc, false, false);
        }
        v8f cacc = a0 + a1;
        // C element v = (row M = v + 8*half, s = wave*16 + N), N = lane%16. Scatter f16 into aFrag.
        unsigned short* afragh = (unsigned short*)sAfrag;
        const int N  = m16;
        const int t  = wave >> 1;
        const int hB = (N >> 3) & 1;
        const int vB = (wave & 1) * 4 + ((N & 7) >> 1);
        const int pr = N & 1;
        #pragma unroll
        for (int v = 0; v < 8; ++v) {
            int rowM = v + 8 * half;
            int idx16 = ((t * 32 + (hB * 16 + rowM)) * 8 + vB) * 2 + pr;
            afragh[idx16] = f16_bits(cacc[v]);
        }
    }
    __syncthreads();

    const float* kh = k + (size_t)head * KLEN * HD;
    unsigned short* bfragh = (unsigned short*)sBfrag;
    const float SCALE = 1.2533141373155003f;             // sqrt(pi/2)

    // Software pipeline: raw k rows for the next chunk are loaded before the est phase.
    float4 raw[16];
    load_raw_row(kh + (size_t)(part * (CHUNKS * KC) + wave * 16 + m16) * HD, half, raw);

    for (int c = 0; c < CHUNKS; ++c) {
        const int k0 = part * (CHUNKS * KC) + c * KC;

        // ---------- Projection: proj = k_tile @ G via 3-term bf16 split WMMA ----------
        unsigned int Ahi[4][8], Alo[4][8];
        cvt_raw(raw, Ahi, Alo);

        for (int stile = 0; stile < 16; ++stile) {
            const int s0 = stile * 16;
            const int scol = s0 + m16;
            v8f a0 = {}, a1 = {};
            #pragma unroll
            for (int st = 0; st < 4; ++st) {
                union { unsigned int u[8]; v16bf v; } bh, bl, ah, al;
                #pragma unroll
                for (int v = 0; v < 8; ++v) {
                    int d = 32 * st + kpos(v, half);
                    bh.u[v] = *(const unsigned int*)(sGthi + scol * GT_STRIDE + d);
                    bl.u[v] = *(const unsigned int*)(sGtlo + scol * GT_STRIDE + d);
                    ah.u[v] = Ahi[st][v];
                    al.u[v] = Alo[st][v];
                }
                v8f& acc = (st & 1) ? a1 : a0;
                acc = __builtin_amdgcn_wmma_f32_16x16x32_bf16(false, ah.v, false, bh.v, (short)0, acc, false, false);
                acc = __builtin_amdgcn_wmma_f32_16x16x32_bf16(false, ah.v, false, bl.v, (short)0, acc, false, false);
                acc = __builtin_amdgcn_wmma_f32_16x16x32_bf16(false, al.v, false, bh.v, (short)0, acc, false, false);
            }
            v8f cacc = a0 + a1;
            // Sign extraction -> scatter ±1 f16 into est B-fragment layout.
            const int N  = m16;
            const int t  = s0 >> 5;
            const int hB = (N >> 3) & 1;
            const int vB = (stile & 1) * 4 + ((N & 7) >> 1);
            const int pr = N & 1;
            #pragma unroll
            for (int v = 0; v < 8; ++v) {
                int rowM = v + 8 * half;
                unsigned short sgn = (cacc[v] > 0.0f) ? (unsigned short)0x3C00 : (unsigned short)0xBC00;
                int idx16 = (((wave * 8 + t) * 32) + (hB * 16 + rowM)) * 16 + vB * 2 + pr;
                bfragh[idx16] = sgn;
            }
        }
        __syncthreads();

        // Issue next chunk's k loads now; latency hides behind est WMMAs + next proj LDS loads.
        if (c + 1 < CHUNKS) {
            load_raw_row(kh + (size_t)(k0 + KC + wave * 16 + m16) * HD, half, raw);
        }

        // ---------- est tile: q_sketch[16x256] @ signs[256 x 16] (f16 WMMA, 2 chains) ----------
        {
            v8f e0 = {}, e1 = {};
            #pragma unroll
            for (int t = 0; t < 8; ++t) {
                union { uint4 q4[2]; v16h v; } A, Bm;
                const uint4* ap = (const uint4*)(sAfrag + (t * 32 + lane) * 8);
                A.q4[0] = ap[0]; A.q4[1] = ap[1];
                const uint4* bp = (const uint4*)(sBfrag + ((wave * 8 + t) * 32 + lane) * 8);
                Bm.q4[0] = bp[0]; Bm.q4[1] = bp[1];
                v8f& acc = (t & 1) ? e1 : e0;
                acc = __builtin_amdgcn_wmma_f32_16x16x32_f16(false, A.v, false, Bm.v, (short)0, acc, false, false);
            }
            v8f acc = e0 + e1;
            const int ncol = m16;
            #pragma unroll
            for (int v = 0; v < 8; ++v) {
                int M = v + 8 * half;
                out[((size_t)(head * QLEN + M)) * KLEN + (k0 + wave * 16 + ncol)] = acc[v] * SCALE;
            }
        }
        __syncthreads();   // protect bFrag before next chunk's sign writes
    }
}

extern "C" void kernel_launch(void* const* d_in, const int* in_sizes, int n_in,
                              void* d_out, int out_size, void* d_ws, size_t ws_size,
                              hipStream_t stream) {
    (void)in_sizes; (void)n_in; (void)d_ws; (void)ws_size; (void)out_size;
    const float* q = (const float*)d_in[0];
    const float* k = (const float*)d_in[1];
    const float* G = (const float*)d_in[2];
    float* out = (float*)d_out;
    qjl_sketch_kernel<<<dim3(HEADS * PARTS), dim3(512), 0, stream>>>(q, k, G, out);
}
